// MoeTransformer_79474074845417
// MI455X (gfx1250) — compile-verified, matrix operand order checked by reading, not run on hardware
//
#include <hip/hip_runtime.h>
#include <cstdint>
#include <cstddef>

// ---------------------------------------------------------------------------
// MoE top-1 transformer FFN layer for gfx1250 (MI455X), bf16 WMMA path.
//   S=8192 tokens, M=2048 d_model, E=8 experts, H=8192 ffn hidden, C=2048 cap
// HBM-bound on fp32 weights (~1.07GB => ~46us floor @ 23.3TB/s); GEMMs use
// v_wmma_f32_16x16x32_bf16 with double-buffered LDS (1 barrier / K-step) and
// branchless gathered A-tiles so global loads issue in clauses that overlap
// the WMMA work of the in-flight tile.
// ---------------------------------------------------------------------------

typedef __attribute__((ext_vector_type(16))) __bf16 v16bf;
typedef __attribute__((ext_vector_type(8)))  float  v8f;

constexpr int S_ = 8192;
constexpr int MD = 2048;
constexpr int E_ = 8;
constexpr int H_ = 8192;
constexpr int C_ = 2048;

constexpr int BT  = 128;      // block tile (M and N)
constexpr int KB  = 32;       // K per WMMA step (bf16)
constexpr int LDT = KB + 2;   // padded LDS row stride (elements)

union FragU { unsigned u[8]; v16bf v; };

// Build a 16x32 bf16 WMMA fragment from an LDS tile (row-major, stride LDT).
// Layout per CDNA5 ISA 7.12.2 (16-bit A-matrix 16x32); B uses the same layout
// on a transposed (N-major) LDS tile.
__device__ __forceinline__ v16bf lds_frag(const __bf16* tile_row0, int lane) {
  const int r    = lane & 15;
  const int half = lane >> 4;
  const char* row = (const char*)(tile_row0 + (size_t)r * LDT);
  FragU f;
#pragma unroll
  for (int j = 0; j < 4; ++j) {
    const int k0 = (half ? 8 : 0) + 2 * j;    // V0..V3
    const int k1 = (half ? 24 : 16) + 2 * j;  // V4..V7
    f.u[j]     = *(const unsigned*)(row + 2 * k0);
    f.u[4 + j] = *(const unsigned*)(row + 2 * k1);
  }
  return f.v;
}

__device__ __forceinline__ float gelu_tanh(float v) {
  const float c0 = 0.7978845608028654f;  // sqrt(2/pi)
  const float c1 = 0.044715f;
  return 0.5f * v * (1.0f + tanhf(c0 * (v + c1 * v * v * v)));
}

// ---------------------------------------------------------------------------
// 0) zero d_out (dropped tokens must produce zeros)
// ---------------------------------------------------------------------------
__global__ __launch_bounds__(256)
void moe_zero_kernel(float4* __restrict__ out, int n4) {
  int i = blockIdx.x * 256 + threadIdx.x;
  if (i < n4) out[i] = make_float4(0.f, 0.f, 0.f, 0.f);
}

// ---------------------------------------------------------------------------
// 1) gate: one wave32 per token. logits = x[s] . Wg[:,e], softmax, argmax.
// ---------------------------------------------------------------------------
__global__ __launch_bounds__(256)
void moe_gate_kernel(const float* __restrict__ x, const float* __restrict__ Wg,
                     int* __restrict__ idx, float* __restrict__ prob) {
  const int gtid = blockIdx.x * 256 + threadIdx.x;
  const int tok  = gtid >> 5;
  const int lane = threadIdx.x & 31;
  if (tok >= S_) return;
  const float* xr = x + (size_t)tok * MD;
  float acc[E_];
#pragma unroll
  for (int e = 0; e < E_; ++e) acc[e] = 0.0f;
  for (int m = lane; m < MD; m += 32) {
    const float xv = xr[m];
    const float4* wp = (const float4*)(Wg + (size_t)m * E_);
    const float4 w0 = wp[0];
    const float4 w1 = wp[1];
    acc[0] += xv * w0.x; acc[1] += xv * w0.y;
    acc[2] += xv * w0.z; acc[3] += xv * w0.w;
    acc[4] += xv * w1.x; acc[5] += xv * w1.y;
    acc[6] += xv * w1.z; acc[7] += xv * w1.w;
  }
#pragma unroll
  for (int off = 16; off > 0; off >>= 1) {
#pragma unroll
    for (int e = 0; e < E_; ++e) acc[e] += __shfl_xor(acc[e], off, 32);
  }
  if (lane == 0) {
    float best = acc[0]; int bi = 0;
#pragma unroll
    for (int e = 1; e < E_; ++e) { if (acc[e] > best) { best = acc[e]; bi = e; } }
    float sum = 0.0f;
#pragma unroll
    for (int e = 0; e < E_; ++e) sum += expf(acc[e] - best);
    idx[tok]  = bi;
    prob[tok] = 1.0f / sum;  // softmax prob at argmax: exp(0)/sum
  }
}

// ---------------------------------------------------------------------------
// 2) routing: deterministic per-expert prefix positions (capacity C).
//    One 1024-thread block, 8 tokens/thread, Hillis-Steele block scan.
//    slotmap[e*C + pos] = token (or -1 empty); gatew[s] = prob * keep.
// ---------------------------------------------------------------------------
__global__ __launch_bounds__(1024)
void moe_route_kernel(const int* __restrict__ idx, const float* __restrict__ prob,
                      int* __restrict__ slotmap, float* __restrict__ gatew) {
  __shared__ int sc[E_][1024];
  const int t = threadIdx.x;
  for (int i = t; i < E_ * C_; i += 1024) slotmap[i] = -1;

  const int s0 = t * (S_ / 1024);  // 8 contiguous tokens per thread
  int eid[S_ / 1024];
  int lc[E_];
#pragma unroll
  for (int e = 0; e < E_; ++e) lc[e] = 0;
#pragma unroll
  for (int j = 0; j < S_ / 1024; ++j) { eid[j] = idx[s0 + j]; lc[eid[j]]++; }
#pragma unroll
  for (int e = 0; e < E_; ++e) sc[e][t] = lc[e];
  __syncthreads();
  for (int off = 1; off < 1024; off <<= 1) {
    int v[E_];
#pragma unroll
    for (int e = 0; e < E_; ++e) v[e] = (t >= off) ? sc[e][t - off] : 0;
    __syncthreads();
#pragma unroll
    for (int e = 0; e < E_; ++e) sc[e][t] += v[e];
    __syncthreads();
  }
  int run[E_];
#pragma unroll
  for (int e = 0; e < E_; ++e) run[e] = sc[e][t] - lc[e];  // exclusive base
#pragma unroll
  for (int j = 0; j < S_ / 1024; ++j) {
    const int s = s0 + j, e = eid[j];
    const int p = run[e]++;
    if (p < C_) { slotmap[e * C_ + p] = s; gatew[s] = prob[s]; }
    else        { gatew[s] = 0.0f; }
  }
}

// ---------------------------------------------------------------------------
// 3) ffn1 (per expert): h[c, n] = gelu( sum_k A[c,k] * W1e[k,n] ), K = MD
//    A rows gathered from x via slotmap (fp32 -> bf16, branchless mask),
//    result bf16 in hbuf. Double-buffered LDS, one barrier per K-step.
// ---------------------------------------------------------------------------
__global__ __launch_bounds__(256)
void moe_ffn1_kernel(const float* __restrict__ x, const float* __restrict__ W1e,
                     const int* __restrict__ slotmap_e, __bf16* __restrict__ hbuf) {
  __shared__ __bf16 As[2][BT][LDT];   // [buf][row c][k]
  __shared__ __bf16 Bs[2][BT][LDT];   // [buf][col n][k] (transposed)
  __shared__ int    stok[BT];

  const int tid  = threadIdx.x;
  const int lane = tid & 31;
  const int wave = tid >> 5;
  const int wm   = wave & 3;          // 4 wave-rows of 32
  const int wn   = wave >> 2;         // 2 wave-cols of 64
  const int rowBase = blockIdx.y * BT;
  const int colBase = blockIdx.x * BT;

  if (tid < BT) stok[tid] = slotmap_e[rowBase + tid];

  const v8f vz = {};
  v8f acc[2][4];
#pragma unroll
  for (int a = 0; a < 2; ++a)
#pragma unroll
    for (int b = 0; b < 4; ++b) acc[a][b] = vz;
  __syncthreads();

  const int ar = tid >> 3, aq = tid & 7;    // A: 32 rows/pass, 8 float4 per row
  const int bk = tid >> 5, bc = tid & 31;   // B: 8 k-rows/pass, 32 float4 per row

  constexpr int NT = MD / KB;               // 64 K-steps

  // ---- tile load into registers (branchless gather for A) ----
  float4 aCur[4], bCur[4];
  auto load_tile = [&](int k0, float4 aR[4], float4 bR[4]) {
#pragma unroll
    for (int p = 0; p < 4; ++p) {
      const int r   = ar + p * 32;
      const int tok = stok[r];
      const int trw = tok < 0 ? 0 : tok;            // clamp: always valid row
      const float msk = tok < 0 ? 0.0f : 1.0f;
      float4 v = *(const float4*)(x + (size_t)trw * MD + k0 + aq * 4);
      aR[p] = make_float4(v.x * msk, v.y * msk, v.z * msk, v.w * msk);
    }
#pragma unroll
    for (int p = 0; p < 4; ++p) {
      const int k = bk + p * 8;
      bR[p] = *(const float4*)(W1e + (size_t)(k0 + k) * H_ + colBase + bc * 4);
    }
  };
  auto store_tile = [&](int buf, const float4 aR[4], const float4 bR[4]) {
#pragma unroll
    for (int p = 0; p < 4; ++p) {
      const int r = ar + p * 32;
      As[buf][r][aq * 4 + 0] = (__bf16)aR[p].x;
      As[buf][r][aq * 4 + 1] = (__bf16)aR[p].y;
      As[buf][r][aq * 4 + 2] = (__bf16)aR[p].z;
      As[buf][r][aq * 4 + 3] = (__bf16)aR[p].w;
    }
#pragma unroll
    for (int p = 0; p < 4; ++p) {
      const int k = bk + p * 8;
      Bs[buf][bc * 4 + 0][k] = (__bf16)bR[p].x;
      Bs[buf][bc * 4 + 1][k] = (__bf16)bR[p].y;
      Bs[buf][bc * 4 + 2][k] = (__bf16)bR[p].z;
      Bs[buf][bc * 4 + 3][k] = (__bf16)bR[p].w;
    }
  };

  load_tile(0, aCur, bCur);

  int buf = 0;
  for (int kt = 0; kt < NT; ++kt) {
    const bool hasNext = (kt + 1 < NT);
    float4 aN[4], bN[4];
    if (hasNext) load_tile((kt + 1) * KB, aN, bN);        // issue global loads early
    if (kt + 2 < NT)                                      // warm L2 two tiles ahead
      __builtin_prefetch(W1e + (size_t)((kt + 2) * KB + bk) * H_ + colBase + bc * 4, 0, 1);

    store_tile(buf, aCur, bCur);                          // writes buf; last reads of
    __syncthreads();                                      // buf were 2 steps ago (safe)

    v16bf afr[2], bfr[4];
#pragma unroll
    for (int ti = 0; ti < 2; ++ti) afr[ti] = lds_frag(&As[buf][wm * 32 + ti * 16][0], lane);
#pragma unroll
    for (int tj = 0; tj < 4; ++tj) bfr[tj] = lds_frag(&Bs[buf][wn * 64 + tj * 16][0], lane);
#pragma unroll
    for (int ti = 0; ti < 2; ++ti)
#pragma unroll
      for (int tj = 0; tj < 4; ++tj)
        acc[ti][tj] = __builtin_amdgcn_wmma_f32_16x16x32_bf16(
            false, afr[ti], false, bfr[tj], (short)0, acc[ti][tj], false, false);

    if (hasNext) {
#pragma unroll
      for (int p = 0; p < 4; ++p) { aCur[p] = aN[p]; bCur[p] = bN[p]; }
    }
    buf ^= 1;
  }

  const int half = lane >> 4, nl = lane & 15;
#pragma unroll
  for (int ti = 0; ti < 2; ++ti)
#pragma unroll
    for (int tj = 0; tj < 4; ++tj)
#pragma unroll
      for (int i = 0; i < 8; ++i) {
        const int m_local = wm * 32 + ti * 16 + half * 8 + i;   // C/D layout
        const int col     = colBase + wn * 64 + tj * 16 + nl;
        hbuf[(size_t)(rowBase + m_local) * H_ + col] = (__bf16)gelu_tanh(acc[ti][tj][i]);
      }
}

// ---------------------------------------------------------------------------
// 4) ffn2 (per expert): o[c, n] = sum_k h[c,k] * W2e[k,n], K = H_.
//    Epilogue scatters out[token] = o * gate_w (combine fused).
// ---------------------------------------------------------------------------
__global__ __launch_bounds__(256)
void moe_ffn2_kernel(const __bf16* __restrict__ hbuf, const float* __restrict__ W2e,
                     const int* __restrict__ slotmap_e, const float* __restrict__ gatew,
                     float* __restrict__ out) {
  __shared__ __bf16 As[2][BT][LDT];
  __shared__ __bf16 Bs[2][BT][LDT];
  __shared__ int    stok[BT];
  __shared__ float  sgate[BT];

  const int tid  = threadIdx.x;
  const int lane = tid & 31;
  const int wave = tid >> 5;
  const int wm   = wave & 3;
  const int wn   = wave >> 2;
  const int rowBase = blockIdx.y * BT;
  const int colBase = blockIdx.x * BT;

  if (tid < BT) {
    const int t = slotmap_e[rowBase + tid];
    stok[tid]  = t;
    sgate[tid] = (t >= 0) ? gatew[t] : 0.0f;
  }

  const v8f vz = {};
  v8f acc[2][4];
#pragma unroll
  for (int a = 0; a < 2; ++a)
#pragma unroll
    for (int b = 0; b < 4; ++b) acc[a][b] = vz;
  __syncthreads();

  const int ar = tid >> 2, ag = tid & 3;    // A: 64 rows/pass, 4x(8 bf16) per row
  const int bk = tid >> 5, bc = tid & 31;

  constexpr int NT = H_ / KB;               // 256 K-steps

  uint4  aCur[2];
  float4 bCur[4];
  auto load_tile = [&](int k0, uint4 aR[2], float4 bR[4]) {
#pragma unroll
    for (int p = 0; p < 2; ++p) {
      const int r = ar + p * 64;
      aR[p] = *(const uint4*)(hbuf + (size_t)(rowBase + r) * H_ + k0 + ag * 8);
    }
#pragma unroll
    for (int p = 0; p < 4; ++p) {
      const int k = bk + p * 8;
      bR[p] = *(const float4*)(W2e + (size_t)(k0 + k) * MD + colBase + bc * 4);
    }
  };
  auto store_tile = [&](int buf, const uint4 aR[2], const float4 bR[4]) {
#pragma unroll
    for (int p = 0; p < 2; ++p) {
      const int r = ar + p * 64;
      unsigned* dst = (unsigned*)&As[buf][r][ag * 8];
      dst[0] = aR[p].x; dst[1] = aR[p].y; dst[2] = aR[p].z; dst[3] = aR[p].w;
    }
#pragma unroll
    for (int p = 0; p < 4; ++p) {
      const int k = bk + p * 8;
      Bs[buf][bc * 4 + 0][k] = (__bf16)bR[p].x;
      Bs[buf][bc * 4 + 1][k] = (__bf16)bR[p].y;
      Bs[buf][bc * 4 + 2][k] = (__bf16)bR[p].z;
      Bs[buf][bc * 4 + 3][k] = (__bf16)bR[p].w;
    }
  };

  load_tile(0, aCur, bCur);

  int buf = 0;
  for (int kt = 0; kt < NT; ++kt) {
    const bool hasNext = (kt + 1 < NT);
    uint4  aN[2];
    float4 bN[4];
    if (hasNext) load_tile((kt + 1) * KB, aN, bN);
    if (kt + 2 < NT)
      __builtin_prefetch(W2e + (size_t)((kt + 2) * KB + bk) * MD + colBase + bc * 4, 0, 1);

    store_tile(buf, aCur, bCur);
    __syncthreads();

    v16bf afr[2], bfr[4];
#pragma unroll
    for (int ti = 0; ti < 2; ++ti) afr[ti] = lds_frag(&As[buf][wm * 32 + ti * 16][0], lane);
#pragma unroll
    for (int tj = 0; tj < 4; ++tj) bfr[tj] = lds_frag(&Bs[buf][wn * 64 + tj * 16][0], lane);
#pragma unroll
    for (int ti = 0; ti < 2; ++ti)
#pragma unroll
      for (int tj = 0; tj < 4; ++tj)
        acc[ti][tj] = __builtin_amdgcn_wmma_f32_16x16x32_bf16(
            false, afr[ti], false, bfr[tj], (short)0, acc[ti][tj], false, false);

    if (hasNext) {
#pragma unroll
      for (int p = 0; p < 2; ++p) aCur[p] = aN[p];
#pragma unroll
      for (int p = 0; p < 4; ++p) bCur[p] = bN[p];
    }
    buf ^= 1;
  }

  const int half = lane >> 4, nl = lane & 15;
#pragma unroll
  for (int ti = 0; ti < 2; ++ti)
#pragma unroll
    for (int tj = 0; tj < 4; ++tj)
#pragma unroll
      for (int i = 0; i < 8; ++i) {
        const int m_local = wm * 32 + ti * 16 + half * 8 + i;
        const int tok = stok[m_local];
        if (tok >= 0) {
          const int col = colBase + wn * 64 + tj * 16 + nl;
          out[(size_t)tok * MD + col] = acc[ti][tj][i] * sgate[m_local];
        }
      }
}

// ---------------------------------------------------------------------------
// Launch. Workspace layout: idx[S] | prob[S] | gatew[S] | slotmap[E*C] |
//                           hbuf[C*H] bf16 (reused across experts) ~33.7 MB
// ---------------------------------------------------------------------------
extern "C" void kernel_launch(void* const* d_in, const int* in_sizes, int n_in,
                              void* d_out, int out_size, void* d_ws, size_t ws_size,
                              hipStream_t stream) {
  (void)in_sizes; (void)n_in; (void)out_size; (void)ws_size;
  const float* x  = (const float*)d_in[0];
  const float* Wg = (const float*)d_in[1];
  const float* W1 = (const float*)d_in[2];
  const float* W2 = (const float*)d_in[3];
  float* out = (float*)d_out;

  char* ws = (char*)d_ws;
  int*    idx     = (int*)ws;
  float*  prob    = (float*)(ws + (size_t)S_ * 4);
  float*  gatew   = (float*)(ws + (size_t)S_ * 8);
  int*    slotmap = (int*)(ws + (size_t)S_ * 12);
  __bf16* hbuf    = (__bf16*)(ws + (size_t)S_ * 12 + (size_t)E_ * C_ * 4);

  moe_zero_kernel<<<(S_ * MD / 4) / 256, 256, 0, stream>>>((float4*)out, S_ * MD / 4);
  moe_gate_kernel<<<S_ / 8, 256, 0, stream>>>(x, Wg, idx, prob);
  moe_route_kernel<<<1, 1024, 0, stream>>>(idx, prob, slotmap, gatew);

  for (int e = 0; e < E_; ++e) {
    moe_ffn1_kernel<<<dim3(H_ / BT, C_ / BT), 256, 0, stream>>>(
        x, W1 + (size_t)e * MD * H_, slotmap + e * C_, hbuf);
    moe_ffn2_kernel<<<dim3(MD / BT, C_ / BT), 256, 0, stream>>>(
        hbuf, W2 + (size_t)e * H_ * MD, slotmap + e * C_, gatew, out);
  }
}
